// EquivariantAttention_9302899163634
// MI455X (gfx1250) — matrix operand order repaired and checked by reading, not compiled, not run on hardware
//
#include <hip/hip_runtime.h>

typedef __attribute__((ext_vector_type(16))) _Float16 v16h;
typedef __attribute__((ext_vector_type(8)))  float    v8f;
typedef __attribute__((ext_vector_type(4)))  unsigned int u32x4;
typedef __attribute__((ext_vector_type(4)))  int i32x4;
typedef __attribute__((ext_vector_type(8)))  int i32x8;

union V16 { v16h h; u32x4 q[2]; };

constexpr int B_ = 2, N_ = 2048, D_ = 256, H_ = 4, DH_ = 64, J_ = 32;
// EIN = 137 -> K padded to 160, LDS row stride 168 halves (bank-friendly)
// hidden 274 -> padded to 288, LDS row stride 296 halves
constexpr int EROW = 168;   // sEdge / we1T row stride (halves)
constexpr int HROW = 296;   // sHid / we2T row stride (halves)
constexpr int MROW = 40;    // sMijH row stride (halves), cols 16..39 zero pad
constexpr int AROW = 72;    // sHidA row stride (halves)

// dynamic LDS layout (bytes)
constexpr unsigned SW1_BYTES  = 288 * EROW * 2;           // 96768
constexpr unsigned SW2_BYTES  = 16 * HROW * 2;            // 9472
constexpr unsigned WAVE_BYTES = 16 * EROW * 2             // sEdge 5376
                              + 16 * HROW * 2             // sHid  9472
                              + 16 * MROW * 2             // sMijH 1280
                              + 16 * AROW * 2             // sHidA 2304
                              + 32 * 4 + 32 * 4;          // sSim/sCw 256
constexpr unsigned SMEM_BYTES = SW1_BYTES + SW2_BYTES + 8 * WAVE_BYTES;  // 255744

__device__ inline v8f wmma_f16(v16h a, v16h b, v8f c) {
  return __builtin_amdgcn_wmma_f32_16x16x32_f16(false, a, false, b, (short)0, c, false, false);
}

// TDM: 1-D tile load global -> LDS. nelem = number of 16-bit elements.
__device__ inline void tdm_load_1d(unsigned lds_addr, const void* gsrc, unsigned nelem) {
  unsigned long long ga = (unsigned long long)(size_t)gsrc;
  u32x4 g0;
  g0[0] = 1u;                                              // count=1, user mode
  g0[1] = lds_addr;                                        // LDS byte address
  g0[2] = (unsigned)(ga & 0xffffffffull);                  // global_addr[31:0]
  g0[3] = (unsigned)((ga >> 32) & 0x01ffffffull) | (2u << 30);  // addr[56:32] | type=2
  i32x8 g1;
  g1[0] = (int)(1u << 16);                                 // data_size = 1 (2 bytes)
  g1[1] = (int)((nelem & 0xffffu) << 16);                  // tensor_dim0[15:0] @ bits63:48
  g1[2] = (int)(((nelem >> 16) & 0xffffu) | (1u << 16));   // dim0 hi | tensor_dim1 = 1
  g1[3] = (int)((nelem & 0xffffu) << 16);                  // tile_dim0 @ bits127:112
  g1[4] = 0;                                               // tile_dim1 = 0 (unused)
  g1[5] = (int)nelem;                                      // tensor_dim0_stride lo
  g1[6] = 0;
  g1[7] = 0;
  i32x4 z4 = {0, 0, 0, 0};
#if defined(__clang_major__) && (__clang_major__ >= 23)
  i32x8 z8 = {0, 0, 0, 0, 0, 0, 0, 0};
  __builtin_amdgcn_tensor_load_to_lds(g0, g1, z4, z4, z8, 0);
#else
  __builtin_amdgcn_tensor_load_to_lds(g0, g1, z4, z4, 0);
#endif
}

// async copy of 128 contiguous bytes global -> LDS (per-lane), ASYNCcnt tracked
__device__ inline void async_copy_row128(void* lds_dst, const void* gsrc) {
  unsigned d = (unsigned)(size_t)lds_dst;                  // low 32 bits = LDS offset
  unsigned long long s = (unsigned long long)(size_t)gsrc;
  asm volatile(
      "global_load_async_to_lds_b128 %0, %1, off\n"
      "global_load_async_to_lds_b128 %0, %1, off offset:16\n"
      "global_load_async_to_lds_b128 %0, %1, off offset:32\n"
      "global_load_async_to_lds_b128 %0, %1, off offset:48\n"
      "global_load_async_to_lds_b128 %0, %1, off offset:64\n"
      "global_load_async_to_lds_b128 %0, %1, off offset:80\n"
      "global_load_async_to_lds_b128 %0, %1, off offset:96\n"
      "global_load_async_to_lds_b128 %0, %1, off offset:112\n"
      :: "v"(d), "v"(s) : "memory");
}

// ---------------------------------------------------------------- prep kernels
__global__ void k_cvt(const float* __restrict__ src, _Float16* __restrict__ dst, int n) {
  for (int i = blockIdx.x * blockDim.x + threadIdx.x; i < n; i += gridDim.x * blockDim.x)
    dst[i] = (_Float16)src[i];
}

// dst[n*KP + k] = (n<Ncols && k<K) ? src[k*Ncols + n] : 0   (transpose + zero pad)
__global__ void k_tpad(const float* __restrict__ src, _Float16* __restrict__ dst,
                       int K, int Ncols, int KP, int NP) {
  int total = KP * NP;
  for (int t = blockIdx.x * blockDim.x + threadIdx.x; t < total; t += gridDim.x * blockDim.x) {
    int nn = t / KP, kk = t % KP;
    float v = (nn < Ncols && kk < K) ? src[kk * Ncols + nn] : 0.f;
    dst[t] = (_Float16)v;
  }
}

// ---------------------------------------------------------------- qkv = feats @ w_qkv
__global__ __launch_bounds__(256) void k_qkv(const _Float16* __restrict__ feats16,
                                             const _Float16* __restrict__ wqkvT,
                                             _Float16* __restrict__ q16,
                                             _Float16* __restrict__ k16,
                                             _Float16* __restrict__ v16) {
  const int lane = threadIdx.x & 31;
  const int wid  = blockIdx.x * 8 + (threadIdx.x >> 5);   // 12288 tiles
  const int rt = wid / 48, ct = wid % 48;
  const int rowA  = rt * 16 + (lane & 15);
  const int ksubA = (lane >> 4) * 8;
  const int colB  = ct * 16 + (lane & 15);
  const int ksubB = (lane >> 4) * 8;
  const u32x4* ap = reinterpret_cast<const u32x4*>(feats16 + (size_t)rowA * 256);
  const u32x4* bp = reinterpret_cast<const u32x4*>(wqkvT  + (size_t)colB * 256);
  v8f acc;
  #pragma unroll
  for (int i = 0; i < 8; ++i) acc[i] = 0.f;
  #pragma unroll
  for (int ks = 0; ks < 8; ++ks) {
    V16 va, vb;
    int a0 = (ks * 32 + ksubA) >> 3;
    va.q[0] = ap[a0]; va.q[1] = ap[a0 + 2];
    int b0 = (ks * 32 + ksubB) >> 3;
    vb.q[0] = bp[b0]; vb.q[1] = bp[b0 + 2];
    acc = wmma_f16(va.h, vb.h, acc);
  }
  const int sel = colB >> 8;
  const int hh  = (colB >> 6) & 3;
  const int dh  = colB & 63;
  _Float16* dst = sel == 0 ? q16 : (sel == 1 ? k16 : v16);
  const int m0 = (lane >> 4) * 8;
  #pragma unroll
  for (int i = 0; i < 8; ++i) {
    int row = rt * 16 + m0 + i;
    int b = row >> 11, nn = row & (N_ - 1);
    dst[(((size_t)b * H_ + hh) * N_ + nn) * 64 + dh] = (_Float16)acc[i];
  }
}

// ---------------------------------------------------------------- fused edge MLP / attention
__global__ __launch_bounds__(256) void k_edge(
    const float* __restrict__ coors, const float* __restrict__ basis,
    const int* __restrict__ nbhd,
    const _Float16* __restrict__ q16, const _Float16* __restrict__ k16,
    const _Float16* __restrict__ v16,
    const _Float16* __restrict__ we1T, const _Float16* __restrict__ we2T,
    const _Float16* __restrict__ wa1T, const _Float16* __restrict__ wc1T,
    const _Float16* __restrict__ wa2T, const _Float16* __restrict__ wc2T,
    const float* __restrict__ b_e1, const float* __restrict__ b_e2,
    const float* __restrict__ b_a1, const float* __restrict__ b_a2,
    const float* __restrict__ b_c1, const float* __restrict__ b_c2,
    _Float16* __restrict__ outh16, float* __restrict__ cpart) {
  extern __shared__ __align__(16) char smem[];
  _Float16* sWe1 = (_Float16*)smem;                       // [288][EROW]
  _Float16* sWe2 = (_Float16*)(smem + SW1_BYTES);         // [16][HROW]
  const int lane = threadIdx.x & 31;
  const int w    = threadIdx.x >> 5;
  char* wb = smem + SW1_BYTES + SW2_BYTES + (size_t)w * WAVE_BYTES;
  _Float16* sEdge = (_Float16*)wb;                        // [16][EROW]
  _Float16* sHid  = (_Float16*)(wb + 16 * EROW * 2);      // [16][HROW]
  _Float16* sMijH = (_Float16*)(wb + 16 * (EROW + HROW) * 2);            // [16][MROW]
  _Float16* sHidA = (_Float16*)(wb + 16 * (EROW + HROW + MROW) * 2);     // [16][AROW]
  float*    sSim  = (float*)(wb + 16 * (EROW + HROW + MROW + AROW) * 2); // [32]
  float*    sCw   = sSim + 32;                                           // [32]

  const int r     = lane & 15;
  const int ksubA = (lane >> 4) * 8;
  const int ksubB = (lane >> 4) * 8;
  const int m0    = (lane >> 4) * 8;
  const int cb16  = lane & 15;

  // ---- stage both edge-MLP weight matrices into LDS once per block via TDM
  if (threadIdx.x < 32) {
    tdm_load_1d((unsigned)(size_t)sWe1, we1T, 288 * EROW);
    tdm_load_1d((unsigned)(size_t)sWe2, we2T, 16 * HROW);
    __builtin_amdgcn_s_wait_tensorcnt((short)0);
  }
  // zero the K-pad columns of the m_ij tile once (they stay zero)
  if (lane < 16) {
    #pragma unroll
    for (int c = 16; c < MROW; ++c) sMijH[r * MROW + c] = (_Float16)0.f;
  }
  __syncthreads();

  const int gw = blockIdx.x * 8 + w;        // (b*H + h)*N + n
  const int n  = gw & (N_ - 1);
  const int bh = gw >> 11;                  // b*H + h
  const int b  = bh >> 2;
  const int h  = bh & 3;
  const size_t qkbase = ((size_t)bh * N_ + n) * 64;
  const int* nrow = nbhd + ((size_t)b * N_ + n) * J_;
  const float* ca = coors + ((size_t)b * N_ + n) * 3;
  const float cax = ca[0], cay = ca[1], caz = ca[2];

  for (int jt = 0; jt < 2; ++jt) {
    const int j   = jt * 16 + r;
    const int idx = nrow[j];
    if (lane < 16) {
      // row r: q (cols 0..63 via async DMA), fourier enc (128..136), zero pad
      _Float16* er = &sEdge[r * EROW];
      async_copy_row128(er, q16 + qkbase);
      const float* cbp = coors + ((size_t)b * N_ + idx) * 3;
      float dx = cax - cbp[0], dy = cay - cbp[1], dz = caz - cbp[2];
      float dist = dx * dx + dy * dy + dz * dz;
      #pragma unroll
      for (int s = 0; s < 4; ++s) {
        float xs = dist * (1.0f / (float)(1 << s));
        er[128 + s] = (_Float16)__sinf(xs);
        er[132 + s] = (_Float16)__cosf(xs);
      }
      er[136] = (_Float16)dist;
      #pragma unroll
      for (int c = 137; c < EROW; ++c) er[c] = (_Float16)0.f;
    } else {
      // row r: gathered k (cols 64..127) via async DMA
      async_copy_row128(&sEdge[r * EROW + 64], k16 + ((size_t)bh * N_ + idx) * 64);
    }
    asm volatile("s_wait_asynccnt 0x0" ::: "memory");
    __syncthreads();

    // GEMM1: edge(16x160) @ we1 -> hid(16x288), bias + relu.
    // A operand is invariant across the 18 column tiles: hoist into registers.
    V16 va[5];
    #pragma unroll
    for (int ks2 = 0; ks2 < 5; ++ks2) {
      const int ka = ks2 * 32 + ksubA;
      va[ks2].q[0] = *reinterpret_cast<const u32x4*>(&sEdge[r * EROW + ka]);
      va[ks2].q[1] = *reinterpret_cast<const u32x4*>(&sEdge[r * EROW + ka + 16]);
    }
    for (int ct = 0; ct < 18; ++ct) {
      v8f acc;
      #pragma unroll
      for (int i = 0; i < 8; ++i) acc[i] = 0.f;
      const int ncol = ct * 16 + cb16;
      const _Float16* brow = sWe1 + (size_t)ncol * EROW;
      #pragma unroll
      for (int ks2 = 0; ks2 < 5; ++ks2) {
        V16 vbv;
        const int kb = ks2 * 32 + ksubB;
        vbv.q[0] = *reinterpret_cast<const u32x4*>(brow + kb);
        vbv.q[1] = *reinterpret_cast<const u32x4*>(brow + kb + 16);
        acc = wmma_f16(va[ks2].h, vbv.h, acc);
      }
      const float bias = (ncol < 274) ? b_e1[ncol] : 0.f;
      #pragma unroll
      for (int i = 0; i < 8; ++i) {
        float v = acc[i] + bias;
        sHid[(m0 + i) * HROW + ncol] = (_Float16)(v > 0.f ? v : 0.f);
      }
    }
    __syncthreads();

    // GEMM2: hid(16x288) @ we2 -> m_ij(16x16) f16, bias + relu
    {
      v8f acc;
      #pragma unroll
      for (int i = 0; i < 8; ++i) acc[i] = 0.f;
      const _Float16* brow = sWe2 + (size_t)cb16 * HROW;
      #pragma unroll
      for (int ks2 = 0; ks2 < 9; ++ks2) {
        V16 vav, vbv;
        const int ka = ks2 * 32 + ksubA;
        vav.q[0] = *reinterpret_cast<const u32x4*>(&sHid[r * HROW + ka]);
        vav.q[1] = *reinterpret_cast<const u32x4*>(&sHid[r * HROW + ka + 16]);
        const int kb = ks2 * 32 + ksubB;
        vbv.q[0] = *reinterpret_cast<const u32x4*>(brow + kb);
        vbv.q[1] = *reinterpret_cast<const u32x4*>(brow + kb + 16);
        acc = wmma_f16(vav.h, vbv.h, acc);
      }
      const float bias = b_e2[cb16];
      #pragma unroll
      for (int i = 0; i < 8; ++i) {
        float v = acc[i] + bias;
        sMijH[(m0 + i) * MROW + cb16] = (_Float16)(v > 0.f ? v : 0.f);
      }
    }
    __syncthreads();

    // heads on WMMA: m_ij(16x16, K padded to 32) -> 64 hidden -> 1 scalar, x2 branches
    #pragma unroll
    for (int br = 0; br < 2; ++br) {
      const _Float16* w1T = br ? wc1T : wa1T;   // [64][32]
      const _Float16* w2T = br ? wc2T : wa2T;   // [16][64], row 0 = w2, rest zero
      const float*    bb1 = br ? b_c1 : b_a1;
      const float     b2v = br ? b_c2[0] : b_a2[0];
      V16 vam;
      vam.q[0] = *reinterpret_cast<const u32x4*>(&sMijH[r * MROW + ksubA]);
      vam.q[1] = *reinterpret_cast<const u32x4*>(&sMijH[r * MROW + ksubA + 16]);
      // stage 1: 4 column tiles, single k-step each (K=32)
      #pragma unroll
      for (int ct = 0; ct < 4; ++ct) {
        v8f acc;
        #pragma unroll
        for (int i = 0; i < 8; ++i) acc[i] = 0.f;
        const int ncol = ct * 16 + cb16;
        const _Float16* brow = w1T + (size_t)ncol * 32;
        V16 vbv;
        vbv.q[0] = *reinterpret_cast<const u32x4*>(brow + ksubB);
        vbv.q[1] = *reinterpret_cast<const u32x4*>(brow + ksubB + 16);
        acc = wmma_f16(vam.h, vbv.h, acc);
        const float bias = bb1[ncol];
        #pragma unroll
        for (int i = 0; i < 8; ++i) {
          float v = acc[i] + bias;
          sHidA[(m0 + i) * AROW + ncol] = (_Float16)(v > 0.f ? v : 0.f);
        }
      }
      asm volatile("" ::: "memory");
      // stage 2: hidden(16x64) @ w2 (as column 0 of 64x16) -> col 0 = result
      v8f acc;
      #pragma unroll
      for (int i = 0; i < 8; ++i) acc[i] = 0.f;
      const _Float16* brow2 = w2T + (size_t)cb16 * 64;
      #pragma unroll
      for (int ks2 = 0; ks2 < 2; ++ks2) {
        V16 vav, vbv;
        const int ka = ks2 * 32 + ksubA;
        vav.q[0] = *reinterpret_cast<const u32x4*>(&sHidA[r * AROW + ka]);
        vav.q[1] = *reinterpret_cast<const u32x4*>(&sHidA[r * AROW + ka + 16]);
        const int kb = ks2 * 32 + ksubB;
        vbv.q[0] = *reinterpret_cast<const u32x4*>(brow2 + kb);
        vbv.q[1] = *reinterpret_cast<const u32x4*>(brow2 + kb + 16);
        acc = wmma_f16(vav.h, vbv.h, acc);
      }
      if (cb16 == 0) {   // lanes 0 and 16 hold rows m0..m0+7 of column 0
        float* dstp = br ? sCw : sSim;
        #pragma unroll
        for (int i = 0; i < 8; ++i) dstp[jt * 16 + m0 + i] = acc[i] + b2v;
      }
      asm volatile("" ::: "memory");
    }
    __syncthreads();
  }

  // softmax over J=32 neighbors (wave32 shuffle reductions, fixed order)
  float s = sSim[lane];
  float mx = s;
  #pragma unroll
  for (int o = 16; o > 0; o >>= 1) mx = fmaxf(mx, __shfl_xor(mx, o, 32));
  float e = __expf(s - mx);
  float se = e;
  #pragma unroll
  for (int o = 16; o > 0; o >>= 1) se += __shfl_xor(se, o, 32);
  const float attn = e / se;
  sSim[lane] = attn;
  __syncthreads();

  // out_h[dh] = sum_j attn_j * v_nb[j][dh]   (lane covers dh and dh+32)
  float o0 = 0.f, o1 = 0.f;
  for (int jj = 0; jj < J_; ++jj) {
    const float a = sSim[jj];
    const int idxj = nrow[jj];
    const _Float16* vr = v16 + ((size_t)bh * N_ + idxj) * 64;
    o0 += a * (float)vr[lane];
    o1 += a * (float)vr[lane + 32];
  }
  _Float16* od = outh16 + ((size_t)b * N_ + n) * 256 + h * 64;
  od[lane]      = (_Float16)o0;
  od[lane + 32] = (_Float16)o1;

  // per-h coordinate partials: cp[c] = sum_j coor_w[j] * basis_nb[j][c]
  {
    const float cw = sCw[lane];
    const int idxj = nrow[lane];
    const float* bs = basis + (((size_t)b * N_ + n) * N_ + idxj) * 3;
    float p0 = cw * bs[0], p1 = cw * bs[1], p2 = cw * bs[2];
    #pragma unroll
    for (int o = 16; o > 0; o >>= 1) {
      p0 += __shfl_xor(p0, o, 32);
      p1 += __shfl_xor(p1, o, 32);
      p2 += __shfl_xor(p2, o, 32);
    }
    if (lane == 0) {
      float* cp = cpart + ((size_t)bh * N_ + n) * 3;
      cp[0] = p0; cp[1] = p1; cp[2] = p2;
    }
  }
}

// ---------------------------------------------------------------- out = outh @ w_out + b_out
__global__ __launch_bounds__(256) void k_outproj(const _Float16* __restrict__ outh16,
                                                 const _Float16* __restrict__ woutT,
                                                 const float* __restrict__ b_out,
                                                 float* __restrict__ out) {
  const int lane = threadIdx.x & 31;
  const int wid  = blockIdx.x * 8 + (threadIdx.x >> 5);   // 4096 tiles
  const int rt = wid >> 4, ct = wid & 15;
  const int rowA  = rt * 16 + (lane & 15);
  const int ksubA = (lane >> 4) * 8;
  const int colB  = ct * 16 + (lane & 15);
  const int ksubB = (lane >> 4) * 8;
  const u32x4* ap = reinterpret_cast<const u32x4*>(outh16 + (size_t)rowA * 256);
  const u32x4* bp = reinterpret_cast<const u32x4*>(woutT  + (size_t)colB * 256);
  v8f acc;
  #pragma unroll
  for (int i = 0; i < 8; ++i) acc[i] = 0.f;
  #pragma unroll
  for (int ks = 0; ks < 8; ++ks) {
    V16 va, vb;
    int a0 = (ks * 32 + ksubA) >> 3;
    va.q[0] = ap[a0]; va.q[1] = ap[a0 + 2];
    int b0 = (ks * 32 + ksubB) >> 3;
    vb.q[0] = bp[b0]; vb.q[1] = bp[b0 + 2];
    acc = wmma_f16(va.h, vb.h, acc);
  }
  const float bias = b_out[colB];
  const int m0 = (lane >> 4) * 8;
  #pragma unroll
  for (int i = 0; i < 8; ++i) {
    int row = rt * 16 + m0 + i;
    out[(size_t)row * 256 + colB] = acc[i] + bias;
  }
}

// ---------------------------------------------------------------- coords: fixed-order h-sum
__global__ void k_coors(const float* __restrict__ cpart, float* __restrict__ outc) {
  int t = blockIdx.x * blockDim.x + threadIdx.x;
  if (t >= B_ * N_ * 3) return;
  int c = t % 3, bn = t / 3;
  int b = bn >> 11, nn = bn & (N_ - 1);
  float s = 0.f;
  for (int h = 0; h < H_; ++h)
    s += cpart[(((size_t)b * H_ + h) * N_ + nn) * 3 + c];
  outc[t] = s;
}

// ---------------------------------------------------------------- launch
extern "C" void kernel_launch(void* const* d_in, const int* in_sizes, int n_in,
                              void* d_out, int out_size, void* d_ws, size_t ws_size,
                              hipStream_t stream) {
  const float* feats = (const float*)d_in[0];
  const float* coors = (const float*)d_in[1];
  const float* basis = (const float*)d_in[2];
  const int*   nbhd  = (const int*)d_in[3];
  const float* w_qkv = (const float*)d_in[4];
  const float* w_out = (const float*)d_in[5];
  const float* b_out = (const float*)d_in[6];
  const float* w_e1  = (const float*)d_in[7];
  const float* b_e1  = (const float*)d_in[8];
  const float* w_e2  = (const float*)d_in[9];
  const float* b_e2  = (const float*)d_in[10];
  const float* w_a1  = (const float*)d_in[11];
  const float* b_a1  = (const float*)d_in[12];
  const float* w_a2  = (const float*)d_in[13];
  const float* b_a2  = (const float*)d_in[14];
  const float* w_c1  = (const float*)d_in[15];
  const float* b_c1  = (const float*)d_in[16];
  const float* w_c2  = (const float*)d_in[17];
  const float* b_c2  = (const float*)d_in[18];
  (void)in_sizes; (void)n_in; (void)out_size; (void)ws_size;

  char* ws = (char*)d_ws;
  size_t off = 0;
  auto take = [&](size_t bytes) -> void* {
    void* p = ws + off;
    off = (off + bytes + 255) & ~(size_t)255;
    return p;
  };
  _Float16* feats16 = (_Float16*)take((size_t)B_ * N_ * D_ * 2);
  _Float16* wqkvT   = (_Float16*)take((size_t)768 * 256 * 2);
  _Float16* we1T    = (_Float16*)take((size_t)288 * EROW * 2);
  _Float16* we2T    = (_Float16*)take((size_t)16 * HROW * 2);
  _Float16* woutT   = (_Float16*)take((size_t)256 * 256 * 2);
  _Float16* wa1T    = (_Float16*)take((size_t)64 * 32 * 2);
  _Float16* wc1T    = (_Float16*)take((size_t)64 * 32 * 2);
  _Float16* wa2T    = (_Float16*)take((size_t)16 * 64 * 2);
  _Float16* wc2T    = (_Float16*)take((size_t)16 * 64 * 2);
  _Float16* q16     = (_Float16*)take((size_t)B_ * H_ * N_ * DH_ * 2);
  _Float16* k16     = (_Float16*)take((size_t)B_ * H_ * N_ * DH_ * 2);
  _Float16* v16     = (_Float16*)take((size_t)B_ * H_ * N_ * DH_ * 2);
  _Float16* outh16  = (_Float16*)take((size_t)B_ * N_ * 256 * 2);
  float*    cpart   = (float*)take((size_t)B_ * H_ * N_ * 3 * 4);

  (void)hipFuncSetAttribute((const void*)k_edge,
                            hipFuncAttributeMaxDynamicSharedMemorySize,
                            (int)SMEM_BYTES);

  k_cvt<<<2048, 256, 0, stream>>>(feats, feats16, B_ * N_ * D_);
  k_tpad<<<256, 256, 0, stream>>>(w_qkv, wqkvT, 256, 768, 256, 768);
  k_tpad<<<64, 256, 0, stream>>>(w_e1, we1T, 137, 274, EROW, 288);
  k_tpad<<<32, 256, 0, stream>>>(w_e2, we2T, 274, 16, HROW, 16);
  k_tpad<<<256, 256, 0, stream>>>(w_out, woutT, 256, 256, 256, 256);
  k_tpad<<<4, 256, 0, stream>>>(w_a1, wa1T, 16, 64, 32, 64);
  k_tpad<<<4, 256, 0, stream>>>(w_c1, wc1T, 16, 64, 32, 64);
  k_tpad<<<4, 256, 0, stream>>>(w_a2, wa2T, 64, 1, 64, 16);
  k_tpad<<<4, 256, 0, stream>>>(w_c2, wc2T, 64, 1, 64, 16);

  k_qkv<<<1536, 256, 0, stream>>>(feats16, wqkvT, q16, k16, v16);

  k_edge<<<2048, 256, SMEM_BYTES, stream>>>(coors, basis, nbhd, q16, k16, v16,
                                            we1T, we2T, wa1T, wc1T, wa2T, wc2T,
                                            b_e1, b_e2, b_a1, b_a2, b_c1, b_c2,
                                            outh16, cpart);

  k_outproj<<<512, 256, 0, stream>>>(outh16, woutT, b_out, (float*)d_out);
  k_coors<<<48, 256, 0, stream>>>(cpart, (float*)d_out + (size_t)B_ * N_ * D_);
}
